// PeptideDecoder_74766790688807
// MI455X (gfx1250) — matrix operand order repaired and checked by reading, not compiled, not run on hardware
//
#include <hip/hip_runtime.h>
#include <hip/hip_bf16.h>
#include <math.h>

#define B_ROWS 8192
#define LATENT 64
#define HIDDEN 1024
#define EMBED  256
#define VOCAB  21
#define MAXLEN 50
#define G4     (4*HIDDEN)                 // 4096 gate rows
#define K1TOT  (HIDDEN+EMBED+HIDDEN)      // 2304 = [h_ctx | emb | h1]
#define K2TOT  (2*HIDDEN)                 // 2048 = [h1 | h2]
#define BPAD   72                         // LDS row stride (halfs), bank-conflict pad

#if defined(__has_builtin)
#if __has_builtin(__builtin_amdgcn_global_load_async_to_lds_b128) && \
    __has_builtin(__builtin_amdgcn_s_wait_asynccnt)
#define USE_ASYNC_LDS 1
#endif
#endif
#ifndef USE_ASYNC_LDS
#define USE_ASYNC_LDS 0
#endif

typedef float  f32x4 __attribute__((ext_vector_type(4)));
typedef float  v8f   __attribute__((ext_vector_type(8)));
typedef __bf16 v16bf __attribute__((ext_vector_type(16)));
typedef int    i32x4v __attribute__((vector_size(16)));   // matches builtin param

union FragBF { f32x4 q[2]; v16bf v; };

__device__ __forceinline__ float sigmoid_f(float x){ return 1.0f/(1.0f + __expf(-x)); }
__device__ __forceinline__ float tanh_f(float x){ float e = __expf(-2.0f*x); return (1.0f - e)/(1.0f + e); }

// ---------------------------------------------------------------------------
// One-time weight packing: fp32 -> bf16, concat [W_ih | W_hh] along K,
// zero-pad W_out 21->32 rows, fuse biases, bf16 embedding table.
// ---------------------------------------------------------------------------
__global__ void k_pack_weights(const float* __restrict__ Wih1, const float* __restrict__ Whh1,
                               const float* __restrict__ Wih2, const float* __restrict__ Whh2,
                               const float* __restrict__ Wout,
                               const float* __restrict__ bih1, const float* __restrict__ bhh1,
                               const float* __restrict__ bih2, const float* __restrict__ bhh2,
                               const float* __restrict__ emb,
                               __bf16* __restrict__ W1c, __bf16* __restrict__ W2c,
                               __bf16* __restrict__ Woc,
                               float* __restrict__ b1, float* __restrict__ b2,
                               __bf16* __restrict__ embb)
{
  long idx = (long)blockIdx.x*blockDim.x + threadIdx.x;
  long stride = (long)gridDim.x*blockDim.x;
  const long n1 = (long)G4*K1TOT;
  const long n2 = (long)G4*K2TOT;
  const long n3 = (long)32*HIDDEN;
  for (long i=idx; i<n1; i+=stride){
    int n = (int)(i / K1TOT), k = (int)(i % K1TOT);
    float v = (k < HIDDEN+EMBED) ? Wih1[(long)n*(HIDDEN+EMBED)+k]
                                 : Whh1[(long)n*HIDDEN + (k-(HIDDEN+EMBED))];
    W1c[i] = (__bf16)v;
  }
  for (long i=idx; i<n2; i+=stride){
    int n = (int)(i / K2TOT), k = (int)(i % K2TOT);
    float v = (k < HIDDEN) ? Wih2[(long)n*HIDDEN+k] : Whh2[(long)n*HIDDEN + (k-HIDDEN)];
    W2c[i] = (__bf16)v;
  }
  for (long i=idx; i<n3; i+=stride){
    int n = (int)(i / HIDDEN), k = (int)(i % HIDDEN);
    float v = (n < VOCAB) ? Wout[(long)n*HIDDEN+k] : 0.0f;
    Woc[i] = (__bf16)v;
  }
  for (long i=idx; i<G4; i+=stride){ b1[i] = bih1[i]+bhh1[i]; b2[i] = bih2[i]+bhh2[i]; }
  for (long i=idx; i<(long)VOCAB*EMBED; i+=stride){ embb[i] = (__bf16)emb[i]; }
}

// ---------------------------------------------------------------------------
// Init: h = LN(z @ Wfc^T + bfc). One block (256 thr) per batch row.
// ---------------------------------------------------------------------------
__global__ void k_init_fc_ln(const float* __restrict__ z, const float* __restrict__ Wfc,
                             const float* __restrict__ bfc,
                             const float* __restrict__ lng, const float* __restrict__ lnb,
                             const float* __restrict__ emb,
                             __bf16* __restrict__ A1, __bf16* __restrict__ A2,
                             float* __restrict__ c1, float* __restrict__ c2)
{
  __shared__ float zrow[LATENT];
  __shared__ float red[256];
  const int r = blockIdx.x, tid = threadIdx.x;
  if (tid < LATENT) zrow[tid] = z[(long)r*LATENT + tid];
  __syncthreads();

  float acc[4]; float s = 0.f, s2 = 0.f;
  #pragma unroll
  for (int j=0;j<4;j++){
    int c = tid*4 + j;
    const float* w = Wfc + (long)c*LATENT;
    float a = bfc[c];
    #pragma unroll 8
    for (int k=0;k<LATENT;k++) a += zrow[k]*w[k];
    acc[j] = a; s += a; s2 += a*a;
  }
  red[tid] = s; __syncthreads();
  for (int o=128;o>0;o>>=1){ if (tid<o) red[tid]+=red[tid+o]; __syncthreads(); }
  float mean = red[0]*(1.0f/HIDDEN); __syncthreads();
  red[tid] = s2; __syncthreads();
  for (int o=128;o>0;o>>=1){ if (tid<o) red[tid]+=red[tid+o]; __syncthreads(); }
  float var = red[0]*(1.0f/HIDDEN) - mean*mean;
  float rstd = rsqrtf(var + 1e-5f);

  __bf16* a1 = A1 + (long)r*K1TOT;
  __bf16* a2 = A2 + (long)r*K2TOT;
  #pragma unroll
  for (int j=0;j<4;j++){
    int c = tid*4 + j;
    float hn = (acc[j]-mean)*rstd*lng[c] + lnb[c];
    __bf16 hb = (__bf16)hn;
    a1[c] = hb;                      // per-step context section
    a1[HIDDEN+EMBED + c] = hb;       // h1 section of A1
    a2[c] = hb;                      // h1 section of A2
    a2[HIDDEN + c] = hb;             // h2 section of A2
    c1[(long)r*HIDDEN + c] = 0.0f;
    c2[(long)r*HIDDEN + c] = 0.0f;
  }
  if (tid < EMBED) a1[HIDDEN + tid] = (__bf16)emb[tid];   // emb[tok=0]
}

// ---------------------------------------------------------------------------
// Fused gate-GEMM + LSTM cell.  gates = A[B,KTOT] @ Wcat^T[KTOT,4096] + bias.
// Block = 256 thr = 8 waves (4 M x 2 N); block tile M=128, 32 gate-cols x 4 gates.
// B (weight) tiles are double-buffer staged in LDS via GLOBAL_LOAD_ASYNC_TO_LDS
// (ASYNCcnt) when the toolchain exposes it, else load+ds_store fallback.
// A fragments stream straight from L2. Cell math fused in epilogue.
// ---------------------------------------------------------------------------
template<int KTOT, bool WRITE2>
__global__ __launch_bounds__(256)
void k_gemm_cell(const __bf16* __restrict__ A,      // [B_ROWS, KTOT]
                 const __bf16* __restrict__ W,      // [4096, KTOT]
                 const float*  __restrict__ bias,   // [4096]
                 float* __restrict__ cstate,        // [B_ROWS, HIDDEN]
                 __bf16* __restrict__ hdst1, int ldh1,
                 __bf16* __restrict__ hdst2, int ldh2)
{
  __shared__ __bf16 Bs[2][128*BPAD];                // 2 x 18 KB
  const int tid  = threadIdx.x;
  const int lane = tid & 31;
  const int wave = tid >> 5;
  const int wm = wave >> 1, wn = wave & 1;
  const int nblk = blockIdx.x*32;                   // block's gate-column base
  const int m0   = blockIdx.y*128 + wm*32;
  const int ll = lane & 15;
  const int hi = lane >> 4;

  v8f acc[2][4];
  #pragma unroll
  for (int mt=0;mt<2;mt++)
    #pragma unroll
    for (int g=0;g<4;g++) acc[mt][g] = v8f{};

  const __bf16* Abase0 = A + (long)(m0 + ll)*KTOT;
  const __bf16* Abase1 = A + (long)(m0 + 16 + ll)*KTOT;

  // Stage one 128-row x 64-half B tile: 1024 16B chunks, 4 per thread.
  // chunk c: W-row wr = c>>3 (gate g = wr>>5, col nn = wr&31), k-off = (c&7)*8 halfs.
  auto stageB = [&](int buf, int k0){
    #pragma unroll
    for (int j=0;j<4;j++){
      const int c  = tid*4 + j;
      const int wr = c >> 3;
      const int ko = (c & 7) << 3;
      const int g  = wr >> 5;
      const int nn = wr & 31;
      const __bf16* gp = W + (long)(g*HIDDEN + nblk + nn)*KTOT + k0 + ko;
      __bf16* lp = &Bs[buf][wr*BPAD + ko];
#if USE_ASYNC_LDS
      __builtin_amdgcn_global_load_async_to_lds_b128(
          (__attribute__((address_space(1))) i32x4v*)const_cast<__bf16*>(gp),
          (__attribute__((address_space(3))) i32x4v*)lp,
          0, 0);
#else
      *(f32x4*)lp = *(const f32x4*)gp;
#endif
    }
  };

  stageB(0, 0);
  int buf = 0;
  for (int k0 = 0; k0 < KTOT; k0 += 64){
    const bool more = (k0 + 64) < KTOT;
    if (more) stageB(buf ^ 1, k0 + 64);
#if USE_ASYNC_LDS
    // In-order ASYNCcnt: <=4 outstanding leaves only the just-issued next-buffer
    // loads in flight, i.e. the current buffer is complete. Last iter: drain all.
    if (more) __builtin_amdgcn_s_wait_asynccnt(4);
    else      __builtin_amdgcn_s_wait_asynccnt(0);
#endif
    __syncthreads();

    if (k0 + 128 < KTOT){
      __builtin_prefetch(Abase0 + k0 + 128, 0, 0);
      __builtin_prefetch(Abase1 + k0 + 128, 0, 0);
    }

    #pragma unroll
    for (int ks=0; ks<2; ks++){
      const int kk = k0 + ks*32;
      FragBF a0, a1, b[4];
      // 16-bit A 16x32 layout: lo lanes K{0..7,16..23}, hi lanes K{8..15,24..31}
      a0.q[0] = *(const f32x4*)(Abase0 + kk + hi*8);
      a0.q[1] = *(const f32x4*)(Abase0 + kk + 16 + hi*8);
      a1.q[0] = *(const f32x4*)(Abase1 + kk + hi*8);
      a1.q[1] = *(const f32x4*)(Abase1 + kk + 16 + hi*8);
      // 16-bit B 32x16 layout from LDS: lane n holds K{0..7,16..23}(lo)/K{8..15,24..31}(hi)
      #pragma unroll
      for (int g=0; g<4; g++){
        const __bf16* lb = &Bs[buf][(g*32 + wn*16 + ll)*BPAD + ks*32 + hi*8];
        b[g].q[0] = *(const f32x4*)lb;
        b[g].q[1] = *(const f32x4*)(lb + 16);
      }
      #pragma unroll
      for (int g=0; g<4; g++){
        acc[0][g] = __builtin_amdgcn_wmma_f32_16x16x32_bf16(false, a0.v, false, b[g].v,
                                                            (short)0, acc[0][g], false, false);
        acc[1][g] = __builtin_amdgcn_wmma_f32_16x16x32_bf16(false, a1.v, false, b[g].v,
                                                            (short)0, acc[1][g], false, false);
      }
    }
    __syncthreads();   // all waves done reading Bs[buf] before it is restaged
    buf ^= 1;
  }

  // Epilogue: fused LSTM cell. D layout: VGPR j, lo lanes -> row j, hi -> row 8+j.
  const int ncol = nblk + wn*16 + ll;
  const float bi = bias[ncol];
  const float bf = bias[HIDDEN   + ncol];
  const float bg = bias[2*HIDDEN + ncol];
  const float bo = bias[3*HIDDEN + ncol];
  #pragma unroll
  for (int mt=0;mt<2;mt++){
    #pragma unroll
    for (int j=0;j<8;j++){
      const int row = m0 + mt*16 + hi*8 + j;
      float iv = acc[mt][0][j] + bi;
      float fv = acc[mt][1][j] + bf;
      float gv = acc[mt][2][j] + bg;
      float ov = acc[mt][3][j] + bo;
      const long cidx = (long)row*HIDDEN + ncol;
      float cn = sigmoid_f(fv)*cstate[cidx] + sigmoid_f(iv)*tanh_f(gv);
      cstate[cidx] = cn;
      float h = sigmoid_f(ov)*tanh_f(cn);
      __bf16 hb = (__bf16)h;
      hdst1[(long)row*ldh1 + ncol] = hb;
      if (WRITE2) hdst2[(long)row*ldh2 + ncol] = hb;
    }
  }
}

// ---------------------------------------------------------------------------
// logits = h2 @ Wout^T + b_out (N padded 21->32, via WMMA), argmax -> tok,
// gather emb[tok] into A1, store logits to d_out[b, t, :].
// ---------------------------------------------------------------------------
__global__ __launch_bounds__(256)
void k_logits_argmax(const __bf16* __restrict__ A2,   // h2 lives at cols [HIDDEN,2H)
                     const __bf16* __restrict__ Woc,  // [32, HIDDEN] bf16 (rows 21..31 = 0)
                     const float*  __restrict__ bout, // [21]
                     const __bf16* __restrict__ embb, // [21, 256] bf16
                     float* __restrict__ out,         // [B, 50, 21]
                     __bf16* __restrict__ A1,         // emb section at col HIDDEN
                     int t)
{
  __shared__ float lds[128][33];
  const int lane = threadIdx.x & 31;
  const int wave = threadIdx.x >> 5;
  const int ll = lane & 15, hi = lane >> 4;
  const int m0 = blockIdx.x*128;
  const int mrow = m0 + wave*16;

  v8f acc0 = v8f{}, acc1 = v8f{};
  const __bf16* Ab  = A2 + (long)(mrow + ll)*K2TOT + HIDDEN;
  const __bf16* Wb0 = Woc + (long)(ll)*HIDDEN       + hi*8;
  const __bf16* Wb1 = Woc + (long)(16 + ll)*HIDDEN  + hi*8;
  for (int k0=0; k0<HIDDEN; k0+=32){
    FragBF a, b0, b1;
    a.q[0]  = *(const f32x4*)(Ab + k0 + hi*8);
    a.q[1]  = *(const f32x4*)(Ab + k0 + 16 + hi*8);
    b0.q[0] = *(const f32x4*)(Wb0 + k0);
    b0.q[1] = *(const f32x4*)(Wb0 + k0 + 16);
    b1.q[0] = *(const f32x4*)(Wb1 + k0);
    b1.q[1] = *(const f32x4*)(Wb1 + k0 + 16);
    acc0 = __builtin_amdgcn_wmma_f32_16x16x32_bf16(false, a.v, false, b0.v, (short)0, acc0, false, false);
    acc1 = __builtin_amdgcn_wmma_f32_16x16x32_bf16(false, a.v, false, b1.v, (short)0, acc1, false, false);
  }
  #pragma unroll
  for (int j=0;j<8;j++){
    lds[wave*16 + hi*8 + j][ll]      = acc0[j];
    lds[wave*16 + hi*8 + j][16 + ll] = acc1[j];
  }
  __syncthreads();

  if (threadIdx.x < 128){
    const int rl = threadIdx.x;
    const long row = m0 + rl;
    float best = -3.4e38f; int btok = 0;
    float* o = out + row*((long)MAXLEN*VOCAB) + (long)t*VOCAB;
    #pragma unroll
    for (int v=0; v<VOCAB; v++){
      float lv = lds[rl][v] + bout[v];
      o[v] = lv;
      if (lv > best){ best = lv; btok = v; }
    }
    const f32x4* src = (const f32x4*)(embb + (long)btok*EMBED);
    f32x4*       dst = (f32x4*)(A1 + row*(long)K1TOT + HIDDEN);
    #pragma unroll
    for (int i=0;i<EMBED/8;i++) dst[i] = src[i];
  }
}

// ---------------------------------------------------------------------------
extern "C" void kernel_launch(void* const* d_in, const int* in_sizes, int n_in,
                              void* d_out, int out_size, void* d_ws, size_t ws_size,
                              hipStream_t stream)
{
  (void)in_sizes; (void)n_in; (void)out_size; (void)ws_size;
  const float* z    = (const float*)d_in[0];
  const float* Wfc  = (const float*)d_in[1];
  const float* bfc  = (const float*)d_in[2];
  const float* lng  = (const float*)d_in[3];
  const float* lnb  = (const float*)d_in[4];
  const float* emb  = (const float*)d_in[5];
  const float* Wih1 = (const float*)d_in[6];
  const float* Whh1 = (const float*)d_in[7];
  const float* bih1 = (const float*)d_in[8];
  const float* bhh1 = (const float*)d_in[9];
  const float* Wih2 = (const float*)d_in[10];
  const float* Whh2 = (const float*)d_in[11];
  const float* bih2 = (const float*)d_in[12];
  const float* bhh2 = (const float*)d_in[13];
  const float* Wout = (const float*)d_in[14];
  const float* bout = (const float*)d_in[15];
  // d_in[16] = max_len (device scalar) — constant 50 in the reference.

  char* p = (char*)d_ws;
  auto alloc = [&](size_t bytes)->char* {
    char* r = p; p += (bytes + 255) & ~(size_t)255; return r;
  };
  __bf16* A1   = (__bf16*)alloc((size_t)B_ROWS*K1TOT*2);   // 37.7 MB
  __bf16* A2   = (__bf16*)alloc((size_t)B_ROWS*K2TOT*2);   // 33.5 MB
  float*  c1   = (float*) alloc((size_t)B_ROWS*HIDDEN*4);  // 33.5 MB
  float*  c2   = (float*) alloc((size_t)B_ROWS*HIDDEN*4);  // 33.5 MB
  __bf16* W1c  = (__bf16*)alloc((size_t)G4*K1TOT*2);       // 18.9 MB
  __bf16* W2c  = (__bf16*)alloc((size_t)G4*K2TOT*2);       // 16.8 MB
  __bf16* Woc  = (__bf16*)alloc((size_t)32*HIDDEN*2);
  float*  b1   = (float*) alloc((size_t)G4*4);
  float*  b2   = (float*) alloc((size_t)G4*4);
  __bf16* embb = (__bf16*)alloc((size_t)VOCAB*EMBED*2);

  k_pack_weights<<<1024,256,0,stream>>>(Wih1,Whh1,Wih2,Whh2,Wout,
                                        bih1,bhh1,bih2,bhh2,emb,
                                        W1c,W2c,Woc,b1,b2,embb);
  k_init_fc_ln<<<B_ROWS,256,0,stream>>>(z,Wfc,bfc,lng,lnb,emb,A1,A2,c1,c2);

  dim3 ggrid(HIDDEN/32, B_ROWS/128);   // (32, 64)
  for (int t=0; t<MAXLEN; ++t){
    k_gemm_cell<K1TOT,true ><<<ggrid,256,0,stream>>>(A1, W1c, b1, c1,
                                                     A1 + (HIDDEN+EMBED), K1TOT,
                                                     A2, K2TOT);
    k_gemm_cell<K2TOT,false><<<ggrid,256,0,stream>>>(A2, W2c, b2, c2,
                                                     A2 + HIDDEN, K2TOT,
                                                     ((__bf16*)nullptr), 0);
    k_logits_argmax<<<B_ROWS/128,256,0,stream>>>(A2, Woc, bout, embb,
                                                 (float*)d_out, A1, t);
  }
}
